// StackedRNN_51488067944976
// MI455X (gfx1250) — matrix operand order repaired and checked by reading, not compiled, not run on hardware
//
#include <hip/hip_runtime.h>
#include <hip/hip_bf16.h>

typedef __attribute__((ext_vector_type(16))) __bf16 v16bf;
typedef __attribute__((ext_vector_type(8)))  __bf16 v8bf;
typedef __attribute__((ext_vector_type(8)))  float  v8f;

#define H 1024
#define B 64
#define T 128
#define L 4
#define HB (B * H)          // 65536 elements per [B,H] plane
#define APAD 8              // LDS row pad (bf16 elems) -> bank-conflict-free A reads
#define ASTR (H + APAD)     // padded LDS row stride

// ---------------------------------------------------------------------------
// Build A-matrix fragment (16x32 bf16, one M-row per lane) from LDS.
// Contiguous K: lanes 0-15 hold K=0..15, lanes 16-31 K=16..31 (matches B).
// p already includes the per-lane +16 offset for high lanes.
// ---------------------------------------------------------------------------
__device__ inline v16bf load_a_frag(const __bf16* p) {
    v8bf lo = *reinterpret_cast<const v8bf*>(p);        // K block +0
    v8bf hi = *reinterpret_cast<const v8bf*>(p + 8);    // K block +8
    v16bf r;
#pragma unroll
    for (int e = 0; e < 8; ++e) { r[e] = lo[e]; r[8 + e] = hi[e]; }
    return r;
}

__device__ inline float sigmoidf_fast(float x) {
    return 1.0f / (1.0f + __expf(-x));
}

// ---------------------------------------------------------------------------
// One LSTM cell step for one layer:
//   gates[64,4096] = x@Wi + h@Wh + b ; c,h update fused in epilogue.
// Grid: (16 col-tiles of 64, 4 batch-tiles of 16). Block: 256 thr = 8 waves.
// Wave w -> gate q=w/2, col subpair s0=(w&1)*2 : two 16x16 WMMA tiles.
// x/h tiles staged once per workgroup in LDS (kills the 8x A-redundancy).
// ---------------------------------------------------------------------------
__global__ __launch_bounds__(256, 2)
void lstm_cell(const __bf16* __restrict__ x,     // [B][H] bf16
               const __bf16* __restrict__ hin,   // [B][H] bf16 (read parity)
               const __bf16* __restrict__ WiT,   // [4H][H] bf16 (transposed)
               const __bf16* __restrict__ WhT,   // [4H][H] bf16 (transposed)
               const float*  __restrict__ bias,  // [4H] f32
               float*        __restrict__ c,     // [B][H] f32 (in/out)
               __bf16*       __restrict__ hout,  // [B][H] bf16 (write parity)
               float*        __restrict__ hf32)  // last layer: d_out (2 copies) or null
{
    const int lane = threadIdx.x & 31;
    const int wave = threadIdx.x >> 5;
    const int m0   = blockIdx.y * 16;        // batch row base
    const int n0   = blockIdx.x * 64;        // hidden col base
    const int q    = wave >> 1;              // gate index 0..3 (i,f,g,o)
    const int s0   = (wave & 1) * 2;         // 16-col subtile pair base
    const int half = lane >> 4;              // 0: low lanes, 1: high lanes
    const int lm   = lane & 15;

    // LDS staging of the workgroup's x/h tiles (16 x 1024 bf16 each, padded)
    __shared__ __bf16 xs[16 * ASTR];
    __shared__ __bf16 hs[16 * ASTR];
    {
        // 16*1024 elems per tile; 256 threads x 8 elems -> 8 iterations each
        for (int i = threadIdx.x * 8; i < 16 * H; i += 256 * 8) {
            const int r = i >> 10;           // row 0..15
            const int k = i & (H - 1);       // col, multiple of 8
            *reinterpret_cast<v8bf*>(&xs[r * ASTR + k]) =
                *reinterpret_cast<const v8bf*>(x + (m0 + r) * H + k);
            *reinterpret_cast<v8bf*>(&hs[r * ASTR + k]) =
                *reinterpret_cast<const v8bf*>(hin + (m0 + r) * H + k);
        }
    }

    // B: one gate column per lane, contiguous K (weights pre-transposed)
    const int col0 = q * H + n0 + s0 * 16 + lm;
    const int col1 = col0 + 16;
    const __bf16* wi0 = WiT + (size_t)col0 * H;
    const __bf16* wi1 = WiT + (size_t)col1 * H;
    const __bf16* wh0 = WhT + (size_t)col0 * H;
    const __bf16* wh1 = WhT + (size_t)col1 * H;
    const int boff = half * 16;              // K offset for B fragment
    const int aoff = half * 16;              // K sub-offset for A fragment
    const __bf16* xrow = xs + lm * ASTR + aoff;
    const __bf16* hrow = hs + lm * ASTR + aoff;

    __syncthreads();

    v8f acc0 = {}; v8f acc1 = {};
    for (int kb = 0; kb < H; kb += 32) {
        v16bf ax = load_a_frag(xrow + kb);
        v16bf b0 = *reinterpret_cast<const v16bf*>(wi0 + kb + boff);
        v16bf b1 = *reinterpret_cast<const v16bf*>(wi1 + kb + boff);
        acc0 = __builtin_amdgcn_wmma_f32_16x16x32_bf16(false, ax, false, b0,
                                                       (short)0, acc0, false, false);
        acc1 = __builtin_amdgcn_wmma_f32_16x16x32_bf16(false, ax, false, b1,
                                                       (short)0, acc1, false, false);
        v16bf ah = load_a_frag(hrow + kb);
        v16bf c0 = *reinterpret_cast<const v16bf*>(wh0 + kb + boff);
        v16bf c1 = *reinterpret_cast<const v16bf*>(wh1 + kb + boff);
        acc0 = __builtin_amdgcn_wmma_f32_16x16x32_bf16(false, ah, false, c0,
                                                       (short)0, acc0, false, false);
        acc1 = __builtin_amdgcn_wmma_f32_16x16x32_bf16(false, ah, false, c1,
                                                       (short)0, acc1, false, false);
    }

    // Stage gates (+bias) to LDS: gl[gate][m][col_local]
    __shared__ float gl[4][16][64];
    const float bv0 = bias[col0];
    const float bv1 = bias[col1];
    const int cl0 = s0 * 16 + lm;
    const int cl1 = cl0 + 16;
    // D layout (ISA 7.12.2): VGPR j -> M = j + half*8, N = lane%16
#pragma unroll
    for (int j = 0; j < 8; ++j) {
        gl[q][j + half * 8][cl0] = acc0[j] + bv0;
        gl[q][j + half * 8][cl1] = acc1[j] + bv1;
    }
    __syncthreads();

    // Fused LSTM elementwise: 16x64 block, 4 elements per thread
    for (int e = threadIdx.x; e < 16 * 64; e += 256) {
        const int m  = e >> 6;
        const int cl = e & 63;
        const int idx = (m0 + m) * H + (n0 + cl);
        const float ig = gl[0][m][cl];
        const float fg = gl[1][m][cl];
        const float gg = gl[2][m][cl];
        const float og = gl[3][m][cl];
        const float cold = c[idx];
        const float cnew = sigmoidf_fast(fg) * cold + sigmoidf_fast(ig) * tanhf(gg);
        const float hnew = sigmoidf_fast(og) * tanhf(cnew);
        c[idx]    = cnew;
        hout[idx] = (__bf16)hnew;
        if (hf32) {                     // last layer: output + final h (identical)
            hf32[idx]      = hnew;
            hf32[HB + idx] = hnew;
        }
    }
}

// ---------------------------------------------------------------------------
// Helper kernels: weight transpose+convert, input convert, zero-fill.
// ---------------------------------------------------------------------------
__global__ void transpose_to_bf16(const float* __restrict__ W,  // [L][H][4H]
                                  __bf16* __restrict__ WT,      // [L][4H][H]
                                  int total)                    // L*4H*H
{
    for (int i = blockIdx.x * blockDim.x + threadIdx.x; i < total;
         i += gridDim.x * blockDim.x) {
        const int l = i / (4 * H * H);
        const int r = i % (4 * H * H);
        const int n = r / H;        // 0..4095
        const int k = r % H;        // 0..1023
        WT[i] = (__bf16)W[(size_t)l * 4 * H * H + (size_t)k * 4 * H + n];
    }
}

__global__ void cvt_f32_bf16(const float* __restrict__ src, __bf16* __restrict__ dst,
                             int n) {
    for (int i = blockIdx.x * blockDim.x + threadIdx.x; i < n;
         i += gridDim.x * blockDim.x)
        dst[i] = (__bf16)src[i];
}

__global__ void zero_bf16(__bf16* p, int n) {
    for (int i = blockIdx.x * blockDim.x + threadIdx.x; i < n;
         i += gridDim.x * blockDim.x)
        p[i] = (__bf16)0.0f;
}

__global__ void zero_f32(float* p, int n) {
    for (int i = blockIdx.x * blockDim.x + threadIdx.x; i < n;
         i += gridDim.x * blockDim.x)
        p[i] = 0.0f;
}

// ---------------------------------------------------------------------------
// Workspace layout (bytes):
//   WiT bf16 : [4][4096][1024]          33,554,432
//   WhT bf16 : [4][4096][1024]          33,554,432
//   x   bf16 : [128][64][1024]          16,777,216
//   h   bf16 : [4][2][64][1024] pingpong 1,048,576
//   c   f32  : [3][64][1024] (layer 3's c lives in d_out)  786,432
// total ~81.8 MB
// ---------------------------------------------------------------------------
extern "C" void kernel_launch(void* const* d_in, const int* in_sizes, int n_in,
                              void* d_out, int out_size, void* d_ws, size_t ws_size,
                              hipStream_t stream) {
    const float* inputs = (const float*)d_in[0];   // [T][B][H]
    const float* Wi     = (const float*)d_in[1];   // [L][H][4H]
    const float* Wh     = (const float*)d_in[2];   // [L][H][4H]
    const float* bias   = (const float*)d_in[3];   // [L][4H]
    float* out = (float*)d_out;                    // [3][B][H]

    char* ws = (char*)d_ws;
    __bf16* WiT = (__bf16*)(ws);
    __bf16* WhT = (__bf16*)(ws + 33554432);
    __bf16* xb  = (__bf16*)(ws + 67108864);
    __bf16* hb  = (__bf16*)(ws + 83886080);
    float*  cws = (float*)(ws + 84934656);

    const int wtot = L * 4 * H * H;
    transpose_to_bf16<<<4096, 256, 0, stream>>>(Wi, WiT, wtot);
    transpose_to_bf16<<<4096, 256, 0, stream>>>(Wh, WhT, wtot);
    cvt_f32_bf16<<<4096, 256, 0, stream>>>(inputs, xb, T * HB);

    // zero initial state: h ping-pong buffers and all c
    zero_bf16<<<512, 256, 0, stream>>>(hb, L * 2 * HB);
    zero_f32 <<<512, 256, 0, stream>>>(cws, 3 * HB);
    zero_f32 <<<256, 256, 0, stream>>>(out + 2 * HB, HB);   // layer-3 c in d_out

    const dim3 grid(16, 4), block(256);
    for (int t = 0; t < T; ++t) {
        const int rp = t & 1;        // read parity
        const int wp = rp ^ 1;       // write parity
        for (int l = 0; l < L; ++l) {
            const __bf16* x = (l == 0) ? xb + (size_t)t * HB
                                       : hb + ((size_t)(l - 1) * 2 + wp) * HB;
            const __bf16* hin  = hb + ((size_t)l * 2 + rp) * HB;
            __bf16*       hout = hb + ((size_t)l * 2 + wp) * HB;
            float* c = (l < 3) ? cws + (size_t)l * HB : out + 2 * HB;
            float* hf32 = (l == 3) ? out : nullptr;
            lstm_cell<<<grid, block, 0, stream>>>(
                x, hin,
                WiT + (size_t)l * 4 * H * H,
                WhT + (size_t)l * 4 * H * H,
                bias + (size_t)l * 4 * H,
                c, hout, hf32);
        }
    }
}
